// MultiheadAttention_74689481277618
// MI455X (gfx1250) — compile-verified, matrix operand order checked by reading, not run on hardware
//
#include <hip/hip_runtime.h>
#include <hip/hip_bf16.h>

typedef __attribute__((ext_vector_type(16))) __bf16 v16bf;
typedef __attribute__((ext_vector_type(8)))  __bf16 v8bf;
typedef __attribute__((ext_vector_type(4)))  __bf16 v4bf;
typedef __attribute__((ext_vector_type(8)))  float  v8f;

// D = A(16x32) * B(32x16) + C, bf16 in / f32 acc
static __device__ __forceinline__ v8f wmma_bf16(v16bf a, v16bf b, v8f c) {
  return __builtin_amdgcn_wmma_f32_16x16x32_bf16(false, a, false, b, (short)0, c,
                                                 false, false);
}

// Load one 16x32 bf16 fragment (A-layout; B-layout is the symmetric transpose,
// so loading rows of a row-major N x K matrix gives the B operand for A*W^T).
// Lane L (sub = L>>4): row = row0 + (L&15); K = {8*sub+0..7, 16+8*sub+0..7}.
static __device__ __forceinline__ v16bf load_frag_rm(const __bf16* __restrict__ base,
                                                     int ld, int row0, int k0, int lane) {
  int r   = row0 + (lane & 15);
  int sub = (lane >> 4) & 1;
  const __bf16* p = base + (size_t)r * ld + k0 + 8 * sub;
  v8bf lo = *(const v8bf*)(p);        // global_load_b128
  v8bf hi = *(const v8bf*)(p + 16);   // global_load_b128
  v16bf out;
#pragma unroll
  for (int i = 0; i < 8; ++i) { out[i] = lo[i]; out[i + 8] = hi[i]; }
  return out;
}

// ---------------- fp32 -> bf16 conversion ----------------
__global__ void cvt_f32_bf16(const float4* __restrict__ src, v4bf* __restrict__ dst, int n4) {
  int i = blockIdx.x * blockDim.x + threadIdx.x;
  if (i < n4) {
    float4 f = src[i];
    v4bf o;
    o[0] = (__bf16)f.x; o[1] = (__bf16)f.y; o[2] = (__bf16)f.z; o[3] = (__bf16)f.w;
    dst[i] = o;
  }
}

// ---------------- GEMM: C = A * W^T ----------------
// A: M x K row-major bf16, W: N x K row-major bf16.
// mode 0: out bf16 head-major  (B,H,S,hd):  n -> (h,d), m -> (b,s)
// mode 1: out bf16 head-transposed (B,H,hd,S)   [for V]
// mode 2: out fp32 row-major M x N, += bias[n]
static __device__ __forceinline__ void store_tile(v8f acc, int mode,
    __bf16* __restrict__ outb, float* __restrict__ outf, const float* __restrict__ bias,
    int m0, int n0, int N, int S, int H, int sub, int col) {
#pragma unroll
  for (int v = 0; v < 8; ++v) {
    int m = m0 + v + 8 * sub;
    int n = n0 + col;
    float val = acc[v];
    if (mode == 2) {
      outf[(size_t)m * N + n] = val + bias[n];
    } else {
      int b = m / S, s = m - b * S;
      int h = n >> 6, d = n & 63;           // hd = 64
      size_t hb = (size_t)(b * H + h);
      if (mode == 0) outb[(hb * S + s) * 64 + d] = (__bf16)val;
      else           outb[(hb * 64 + d) * S + s] = (__bf16)val;
    }
  }
}

// Wave tile 32(M) x 64(N): 2 A-frags x 4 B-frags -> 8 WMMAs per 12 b128 loads.
// Block = 128 threads (4 waves) tiling 64 x 128.
__global__ __launch_bounds__(128) void gemm_bf16_wmma(
    const __bf16* __restrict__ A, const __bf16* __restrict__ W,
    int M, int N, int Kdim,
    __bf16* __restrict__ outb, float* __restrict__ outf, const float* __restrict__ bias,
    int mode, int S, int H) {
  int lane = threadIdx.x & 31;
  int wave = threadIdx.x >> 5;
  int sub  = (lane >> 4) & 1;
  int col  = lane & 15;
  int bm = blockIdx.x * 64  + (wave >> 1) * 32;
  int bn = blockIdx.y * 128 + (wave & 1) * 64;

  v8f acc00 = {}, acc01 = {}, acc02 = {}, acc03 = {};
  v8f acc10 = {}, acc11 = {}, acc12 = {}, acc13 = {};
  for (int k0 = 0; k0 < Kdim; k0 += 32) {
    v16bf a0 = load_frag_rm(A, Kdim, bm,      k0, lane);
    v16bf a1 = load_frag_rm(A, Kdim, bm + 16, k0, lane);
    v16bf b0 = load_frag_rm(W, Kdim, bn,      k0, lane);
    v16bf b1 = load_frag_rm(W, Kdim, bn + 16, k0, lane);
    v16bf b2 = load_frag_rm(W, Kdim, bn + 32, k0, lane);
    v16bf b3 = load_frag_rm(W, Kdim, bn + 48, k0, lane);
    acc00 = wmma_bf16(a0, b0, acc00);
    acc01 = wmma_bf16(a0, b1, acc01);
    acc02 = wmma_bf16(a0, b2, acc02);
    acc03 = wmma_bf16(a0, b3, acc03);
    acc10 = wmma_bf16(a1, b0, acc10);
    acc11 = wmma_bf16(a1, b1, acc11);
    acc12 = wmma_bf16(a1, b2, acc12);
    acc13 = wmma_bf16(a1, b3, acc13);
  }
  store_tile(acc00, mode, outb, outf, bias, bm,      bn,      N, S, H, sub, col);
  store_tile(acc01, mode, outb, outf, bias, bm,      bn + 16, N, S, H, sub, col);
  store_tile(acc02, mode, outb, outf, bias, bm,      bn + 32, N, S, H, sub, col);
  store_tile(acc03, mode, outb, outf, bias, bm,      bn + 48, N, S, H, sub, col);
  store_tile(acc10, mode, outb, outf, bias, bm + 16, bn,      N, S, H, sub, col);
  store_tile(acc11, mode, outb, outf, bias, bm + 16, bn + 16, N, S, H, sub, col);
  store_tile(acc12, mode, outb, outf, bias, bm + 16, bn + 32, N, S, H, sub, col);
  store_tile(acc13, mode, outb, outf, bias, bm + 16, bn + 48, N, S, H, sub, col);
}

// ---------------- Flash attention (causal) ----------------
// Q, K: (B*H, S, 64) bf16 row-major.  Vt: (B*H, 64, S) bf16 row-major.
// ctx out: (B, S, H*64) bf16 row-major.
// Block = 128 threads (4 waves); wave w owns query rows qbase..qbase+15.
__global__ __launch_bounds__(128) void attn_flash(
    const __bf16* __restrict__ Q, const __bf16* __restrict__ Km,
    const __bf16* __restrict__ Vt, __bf16* __restrict__ ctx, int S, int H) {
  __shared__ __align__(16) __bf16 pbuf[4][16 * 32];   // per-wave P staging (1 KB each)

  int lane = threadIdx.x & 31;
  int wave = threadIdx.x >> 5;
  int sub  = (lane >> 4) & 1;
  int col  = lane & 15;

  int bh = blockIdx.y;
  int b  = bh / H, h = bh - b * H;
  const __bf16* q  = Q  + (size_t)bh * S * 64;
  const __bf16* k  = Km + (size_t)bh * S * 64;
  const __bf16* vt = Vt + (size_t)bh * 64 * S;

  int qbase = blockIdx.x * 64 + wave * 16;

  v16bf qa0 = load_frag_rm(q, 64, qbase, 0,  lane);   // d 0..31
  v16bf qa1 = load_frag_rm(q, 64, qbase, 32, lane);   // d 32..63

  v8f o0 = {}, o1 = {}, o2 = {}, o3 = {};
  float mrow[8], lrow[8];
#pragma unroll
  for (int v = 0; v < 8; ++v) { mrow[v] = -3.0e38f; lrow[v] = 0.0f; }

  int rowg  = qbase + 8 * sub;                 // global query row for VGPR v is rowg+v
  int qlast = qbase + 15;
  __bf16* pb = &pbuf[wave][0];

  for (int kb = 0; kb <= qlast; kb += 32) {
    // --- scores: S0 = keys kb..kb+15, S1 = keys kb+16..kb+31 ---
    v16bf kb00 = load_frag_rm(k, 64, kb,      0,  lane);
    v16bf kb01 = load_frag_rm(k, 64, kb,      32, lane);
    v16bf kb10 = load_frag_rm(k, 64, kb + 16, 0,  lane);
    v16bf kb11 = load_frag_rm(k, 64, kb + 16, 32, lane);
    v8f s0 = {}, s1 = {};
    s0 = wmma_bf16(qa0, kb00, s0);
    s0 = wmma_bf16(qa1, kb01, s0);
    s1 = wmma_bf16(qa0, kb10, s1);
    s1 = wmma_bf16(qa1, kb11, s1);

    int c0 = kb + col, c1 = kb + 16 + col;
    float rmax[8];
#pragma unroll
    for (int v = 0; v < 8; ++v) {
      float a  = s0[v] * 0.125f;               // 1/sqrt(64)
      float bb = s1[v] * 0.125f;
      a  = (c0 > rowg + v) ? -3.0e38f : a;     // causal mask
      bb = (c1 > rowg + v) ? -3.0e38f : bb;
      s0[v] = a; s1[v] = bb;
      float t = fmaxf(a, bb);                  // reduce across the 16-lane half
      t = fmaxf(t, __shfl_xor(t, 1, 32));
      t = fmaxf(t, __shfl_xor(t, 2, 32));
      t = fmaxf(t, __shfl_xor(t, 4, 32));
      t = fmaxf(t, __shfl_xor(t, 8, 32));
      rmax[v] = t;
    }
#pragma unroll
    for (int v = 0; v < 8; ++v) {
      float mn = fmaxf(mrow[v], rmax[v]);
      float c  = __expf(mrow[v] - mn);
      mrow[v]  = mn;
      float p0 = __expf(s0[v] - mn);
      float p1 = __expf(s1[v] - mn);
      s0[v] = p0; s1[v] = p1;
      float rs = p0 + p1;
      rs += __shfl_xor(rs, 1, 32);
      rs += __shfl_xor(rs, 2, 32);
      rs += __shfl_xor(rs, 4, 32);
      rs += __shfl_xor(rs, 8, 32);
      lrow[v] = lrow[v] * c + rs;
      o0[v] *= c; o1[v] *= c; o2[v] *= c; o3[v] *= c;
    }

    // --- transpose P (C-layout) -> A-layout via per-wave LDS ---
#pragma unroll
    for (int v = 0; v < 8; ++v) {
      pb[(v + 8 * sub) * 32 + col]      = (__bf16)s0[v];
      pb[(v + 8 * sub) * 32 + 16 + col] = (__bf16)s1[v];
    }
    asm volatile("s_wait_dscnt 0" ::: "memory");
    v16bf pa;
    {
      const __bf16* pp = pb + col * 32 + 8 * sub;   // row = query, K = keys
      v8bf lo = *(const v8bf*)(pp);                 // ds_load_b128
      v8bf hi = *(const v8bf*)(pp + 16);
#pragma unroll
      for (int i = 0; i < 8; ++i) { pa[i] = lo[i]; pa[8 + i] = hi[i]; }
    }

    // --- O += P * V   (B operand from Vt rows = head-dim) ---
    v16bf vb0 = load_frag_rm(vt, S, 0,  kb, lane);
    v16bf vb1 = load_frag_rm(vt, S, 16, kb, lane);
    v16bf vb2 = load_frag_rm(vt, S, 32, kb, lane);
    v16bf vb3 = load_frag_rm(vt, S, 48, kb, lane);
    o0 = wmma_bf16(pa, vb0, o0);
    o1 = wmma_bf16(pa, vb1, o1);
    o2 = wmma_bf16(pa, vb2, o2);
    o3 = wmma_bf16(pa, vb3, o3);
  }

  // --- epilogue: normalize and write ctx (B, S, H*64) ---
  int D = H * 64;
#pragma unroll
  for (int v = 0; v < 8; ++v) {
    float inv = 1.0f / lrow[v];
    int srow  = rowg + v;
    size_t base = ((size_t)(b * S + srow)) * D + h * 64 + col;
    ctx[base +  0] = (__bf16)(o0[v] * inv);
    ctx[base + 16] = (__bf16)(o1[v] * inv);
    ctx[base + 32] = (__bf16)(o2[v] * inv);
    ctx[base + 48] = (__bf16)(o3[v] * inv);
  }
}

// ---------------- host launcher ----------------
extern "C" void kernel_launch(void* const* d_in, const int* in_sizes, int n_in,
                              void* d_out, int out_size, void* d_ws, size_t ws_size,
                              hipStream_t stream) {
  const float* x  = (const float*)d_in[0];
  const float* Wq = (const float*)d_in[1];
  const float* Wk = (const float*)d_in[2];
  const float* Wv = (const float*)d_in[3];
  const float* Wo = (const float*)d_in[4];
  const float* bo = (const float*)d_in[5];

  const int B = 2, S = 2048, D = 1024, H = 16;
  const int M = B * S;

  char* ws = (char*)d_ws;
  size_t off = 0;
  __bf16* xb   = (__bf16*)(ws + off); off += (size_t)M * D * 2;
  __bf16* wqb  = (__bf16*)(ws + off); off += (size_t)D * D * 2;
  __bf16* wkb  = (__bf16*)(ws + off); off += (size_t)D * D * 2;
  __bf16* wvb  = (__bf16*)(ws + off); off += (size_t)D * D * 2;
  __bf16* wob  = (__bf16*)(ws + off); off += (size_t)D * D * 2;
  __bf16* Qb   = (__bf16*)(ws + off); off += (size_t)M * D * 2;   // (B,H,S,64)
  __bf16* Kb   = (__bf16*)(ws + off); off += (size_t)M * D * 2;   // (B,H,S,64)
  __bf16* Vtb  = (__bf16*)(ws + off); off += (size_t)M * D * 2;   // (B,H,64,S)
  __bf16* ctxb = (__bf16*)(ws + off); off += (size_t)M * D * 2;   // (B,S,D)

  // fp32 -> bf16 conversions (inputs then stay L2-resident in half the bytes)
  {
    int n4 = (M * D) / 4;
    cvt_f32_bf16<<<(n4 + 255) / 256, 256, 0, stream>>>((const float4*)x, (v4bf*)xb, n4);
    n4 = (D * D) / 4;
    cvt_f32_bf16<<<(n4 + 255) / 256, 256, 0, stream>>>((const float4*)Wq, (v4bf*)wqb, n4);
    cvt_f32_bf16<<<(n4 + 255) / 256, 256, 0, stream>>>((const float4*)Wk, (v4bf*)wkb, n4);
    cvt_f32_bf16<<<(n4 + 255) / 256, 256, 0, stream>>>((const float4*)Wv, (v4bf*)wvb, n4);
    cvt_f32_bf16<<<(n4 + 255) / 256, 256, 0, stream>>>((const float4*)Wo, (v4bf*)wob, n4);
  }

  dim3 gg(M / 64, D / 128);
  // Q, K head-major; V stored transposed per head for contiguous PV B-operand loads
  gemm_bf16_wmma<<<gg, 128, 0, stream>>>(xb, wqb, M, D, D, Qb,  nullptr, nullptr, 0, S, H);
  gemm_bf16_wmma<<<gg, 128, 0, stream>>>(xb, wkb, M, D, D, Kb,  nullptr, nullptr, 0, S, H);
  gemm_bf16_wmma<<<gg, 128, 0, stream>>>(xb, wvb, M, D, D, Vtb, nullptr, nullptr, 1, S, H);

  attn_flash<<<dim3(S / 64, B * H), 128, 0, stream>>>(Qb, Kb, Vtb, ctxb, S, H);

  // output projection: fp32 out + bias
  gemm_bf16_wmma<<<gg, 128, 0, stream>>>(ctxb, wob, M, D, D, nullptr, (float*)d_out, bo, 2, S, H);
}
